// DotProductAttention_52149492908707
// MI455X (gfx1250) — matrix operand order repaired and checked by reading, not compiled, not run on hardware
//
#include <hip/hip_runtime.h>

typedef __attribute__((ext_vector_type(16))) _Float16 v16h;
typedef __attribute__((ext_vector_type(8)))  float    v8f;
typedef __attribute__((ext_vector_type(4)))  unsigned int u32x4;

#define S_LEN 2048
#define DKK   64
#define KC    32            // keys per pipeline step
#define NWAVE 8             // waves per workgroup
#define QROWS (NWAVE * 16)  // q rows per workgroup
#define NIT   (S_LEN / KC)
#define MWRDS (S_LEN / 32)  // packed mask words per row
// padded LDS row strides in halves (multiples of 8 -> 16B-aligned b128 rows)
#define SKP (DKK + 8)       // 72
#define SVP (KC + 8)        // 40
#define SPP (KC + 8)        // 40

#if __has_builtin(__builtin_amdgcn_exp2f)
#define EXP2F(x) __builtin_amdgcn_exp2f(x)   // raw v_exp_f32
#else
#define EXP2F(x) exp2f(x)
#endif

union FragB {
  v16h v;
  unsigned int d[8];
  u32x4 q[2];
};

// one v_cvt_pkrtz_f16_f32: two floats -> packed 2xf16 (as u32)
__device__ __forceinline__ unsigned int f2h2(float lo, float hi) {
  return __builtin_bit_cast(unsigned int, __builtin_amdgcn_cvt_pkrtz(lo, hi));
}

// ---- one-time pre-pass: pack int32 0/1 mask into bitmask words ----
__global__ __launch_bounds__(256)
void pack_mask_kernel(const int* __restrict__ mask, unsigned* __restrict__ mbits) {
  const size_t i = (size_t)blockIdx.x * 256 + threadIdx.x;
  const unsigned long long b = __ballot(mask[i] != 0);
  if ((threadIdx.x & 31) == 0) mbits[i >> 5] = (unsigned)b;
}

template <bool MB>
__global__ __launch_bounds__(256)
void attn_kernel(const float* __restrict__ Q, const float* __restrict__ K,
                 const float* __restrict__ V, const int* __restrict__ maskI,
                 const unsigned* __restrict__ maskW, float* __restrict__ O) {
  __shared__ unsigned short sK[2][KC * SKP];     // K chunk f16 [k][d], double buffered
  __shared__ unsigned short sVt[2][DKK * SVP];   // V chunk f16 [d][perm(k)]
  __shared__ unsigned short sP[NWAVE][16 * SPP]; // per-wave P tile f16 [m][perm(k)]

  const int tid  = threadIdx.x;
  const int lane = tid & 31;
  const int w    = tid >> 5;
  const int ln16 = lane & 15;
  const int sel  = lane >> 4;

  const int bh   = blockIdx.x >> 4;   // B*H = 64
  const int qblk = blockIdx.x & 15;   // 16 q-blocks of 128 rows
  const int q0   = qblk * QROWS + w * 16;

  const float    scale2 = 0.125f * 1.44269504f;       // 1/sqrt(64) * log2(e)
  const unsigned mpat   = __float_as_uint(-1.44269504e9f);

  const float* Qh = Q + (size_t)bh * S_LEN * DKK;
  const float* Kh = K + (size_t)bh * S_LEN * DKK;
  const float* Vh = V + (size_t)bh * S_LEN * DKK;
  float*       Oh = O + (size_t)bh * S_LEN * DKK;

  // ---- persistent A-fragments of Q (16 rows x 64 d, two K=32 chunks) ----
  // A order used here (matches bK): K contiguous per half-wave: K = sel*16 .. sel*16+15
  FragB aQ[2];
  {
    const float* qrow = Qh + (size_t)(q0 + ln16) * DKK;
    #pragma unroll
    for (int c = 0; c < 2; ++c) {
      const int d0 = c * 32 + sel * 16;
      #pragma unroll
      for (int j = 0; j < 4; ++j) aQ[c].d[j]     = f2h2(qrow[d0 + 2*j],      qrow[d0 + 2*j + 1]);
      #pragma unroll
      for (int j = 0; j < 4; ++j) aQ[c].d[4 + j] = f2h2(qrow[d0 + 8 + 2*j], qrow[d0 + 8 + 2*j + 1]);
    }
  }

  v8f acc[4] = {};
  float rowsum[8];                 // per-lane partial, reduced once at the end
  #pragma unroll
  for (int r = 0; r < 8; ++r) rowsum[r] = 0.0f;

  const int*      mrowI[8];
  const unsigned* mrowW[8];
  if (MB) {
    #pragma unroll
    for (int r = 0; r < 8; ++r)
      mrowW[r] = maskW + (size_t)(q0 + r + 8 * sel) * MWRDS;
  } else {
    #pragma unroll
    for (int r = 0; r < 8; ++r)
      mrowI[r] = maskI + (size_t)(q0 + r + 8 * sel) * S_LEN;
  }

  // ---- staging mappings ----
  const int skk = tid >> 3;          // K: row skk, 8 contiguous d at sd0
  const int sd0 = (tid & 7) * 8;
  const int vd  = tid & 63;          // V: column d=vd, logical rows {vg..vg+3, +16}
  const int vg  = (tid >> 6) * 4;    //    -> physical interleaved cols 2*vg..2*vg+7

  float kreg[8], vlo[4], vhi[4];

  auto ld = [&](int kb) {
    const float* kp = Kh + (size_t)(kb + skk) * DKK + sd0;
    #pragma unroll
    for (int j = 0; j < 8; ++j) kreg[j] = kp[j];
    const float* vp = Vh + (size_t)(kb + vg) * DKK + vd;
    #pragma unroll
    for (int i = 0; i < 4; ++i) {
      vlo[i] = vp[(size_t)i * DKK];
      vhi[i] = vp[(size_t)(i + 16) * DKK];
    }
  };
  auto st = [&](int buf) {
    u32x4 pk;
    pk.x = f2h2(kreg[0], kreg[1]);
    pk.y = f2h2(kreg[2], kreg[3]);
    pk.z = f2h2(kreg[4], kreg[5]);
    pk.w = f2h2(kreg[6], kreg[7]);
    *(u32x4*)&sK[buf][skk * SKP + sd0] = pk;
    u32x4 pv;
    pv.x = f2h2(vlo[0], vhi[0]);
    pv.y = f2h2(vlo[1], vhi[1]);
    pv.z = f2h2(vlo[2], vhi[2]);
    pv.w = f2h2(vlo[3], vhi[3]);
    *(u32x4*)&sVt[buf][vd * SVP + 2 * vg] = pv;
  };

  // ---- software pipeline prologue ----
  ld(0);
  st(0);
  ld(KC);

  #pragma clang loop unroll_count(2)
  for (int it = 0; it < NIT; ++it) {
    const int cur = it & 1;
    const int kb  = it * KC;
    __syncthreads();

    if (it + 1 < NIT) st(cur ^ 1);
    if (it + 2 < NIT) ld(kb + 2 * KC);

    // mask for this wave's 16x32 score tile
    unsigned mwd[8];
    int mi0[8], mi1[8];
    if (MB) {
      #pragma unroll
      for (int r = 0; r < 8; ++r) mwd[r] = mrowW[r][it];
    } else {
      #pragma unroll
      for (int r = 0; r < 8; ++r) {
        mi0[r] = mrowI[r][kb + ln16];
        mi1[r] = mrowI[r][kb + 16 + ln16];
      }
    }

    // ---- B-fragments of K^T from LDS ----
    // B order used here (same K order as aQ): lane = column N, K-dim contiguous per half-wave:
    // K = sel*16 .. sel*16+15  (V0=K0,1 ... V7=K14,15 per half)
    FragB bK[2][2];
    #pragma unroll
    for (int t = 0; t < 2; ++t) {
      const int kl = ln16 + 16 * t;
      #pragma unroll
      for (int c = 0; c < 2; ++c) {
        const int d0 = c * 32 + sel * 16;
        bK[c][t].q[0] = *(const u32x4*)&sK[cur][kl * SKP + d0];
        bK[c][t].q[1] = *(const u32x4*)&sK[cur][kl * SKP + d0 + 8];
      }
    }

    // ---- S = Q K^T (16q x 32k), fp32 accumulate over DK=64 ----
    v8f s0 = {}, s1 = {};
    s0 = __builtin_amdgcn_wmma_f32_16x16x32_f16(false, aQ[0].v, false, bK[0][0].v, (short)0, s0, false, false);
    s0 = __builtin_amdgcn_wmma_f32_16x16x32_f16(false, aQ[1].v, false, bK[1][0].v, (short)0, s0, false, false);
    s1 = __builtin_amdgcn_wmma_f32_16x16x32_f16(false, aQ[0].v, false, bK[0][1].v, (short)0, s1, false, false);
    s1 = __builtin_amdgcn_wmma_f32_16x16x32_f16(false, aQ[1].v, false, bK[1][1].v, (short)0, s1, false, false);

    // ---- softmax numerator (log2 domain, no running max: scores bounded;
    //      clamp at 15 keeps P finite in f16) ----
    #pragma unroll
    for (int r = 0; r < 8; ++r) {
      float b0, b1;
      if (MB) {
        const unsigned wd = mwd[r];
        b0 = __uint_as_float(((wd >> ln16) & 1u) ? mpat : 0u);
        b1 = __uint_as_float(((wd >> (ln16 + 16)) & 1u) ? mpat : 0u);
      } else {
        b0 = __uint_as_float((0u - (unsigned)mi0[r]) & mpat);
        b1 = __uint_as_float((0u - (unsigned)mi1[r]) & mpat);
      }
      const float p0 = EXP2F(fminf(s0[r] * scale2 + b0, 15.0f));
      const float p1 = EXP2F(fminf(s1[r] * scale2 + b1, 15.0f));
      rowsum[r] += p0 + p1;
      // logical k and k+16 -> adjacent physical cols 2k, 2k+1
      *(unsigned*)&sP[w][(r + 8 * sel) * SPP + 2 * ln16] = f2h2(p0, p1);
    }

    // ---- A-fragment of P (16 x 32, physical/permuted k; A interleaved layout) ----
    FragB aP;
    aP.q[0] = *(const u32x4*)&sP[w][ln16 * SPP + sel * 8];
    aP.q[1] = *(const u32x4*)&sP[w][ln16 * SPP + sel * 8 + 16];

    // ---- acc += P V (B layout: lane = column n, same physical cols as aP) ----
    #pragma unroll
    for (int t = 0; t < 4; ++t) {
      FragB bV;
      const int n = t * 16 + ln16;
      bV.q[0] = *(const u32x4*)&sVt[cur][n * SVP + sel * 8];
      bV.q[1] = *(const u32x4*)&sVt[cur][n * SVP + sel * 8 + 16];
      acc[t] = __builtin_amdgcn_wmma_f32_16x16x32_f16(false, aP.v, false, bV.v, (short)0, acc[t], false, false);
    }
  }

  // ---- final: reduce per-lane rowsum partials across the 16-lane group, store ----
  #pragma unroll
  for (int r = 0; r < 8; ++r) {
    float s = rowsum[r];
    #pragma unroll
    for (int off = 1; off < 16; off <<= 1)
      s += __shfl_xor(s, off, 32);
    const float inv = 1.0f / s;
    float* orow = Oh + (size_t)(q0 + r + 8 * sel) * DKK;
    #pragma unroll
    for (int t = 0; t < 4; ++t)
      orow[t * 16 + ln16] = acc[t][r] * inv;
  }
}

extern "C" void kernel_launch(void* const* d_in, const int* in_sizes, int n_in,
                              void* d_out, int out_size, void* d_ws, size_t ws_size,
                              hipStream_t stream) {
  (void)in_sizes; (void)n_in; (void)out_size;
  const float* Q    = (const float*)d_in[0];
  const float* K    = (const float*)d_in[1];
  const float* V    = (const float*)d_in[2];
  // d_in[3] is d_k (scalar 64) — folded into scale2
  const int*   mask = (const int*)d_in[4];
  float* O = (float*)d_out;

  const dim3 grid(64 * (S_LEN / QROWS));  // 1024 workgroups
  const dim3 block(256);                  // 8 waves of 32

  const size_t need = (size_t)S_LEN * MWRDS * sizeof(unsigned);  // 512 KB
  if (ws_size >= need) {
    unsigned* mbits = (unsigned*)d_ws;
    hipLaunchKernelGGL(pack_mask_kernel,
                       dim3((S_LEN * (size_t)S_LEN) / 256), dim3(256), 0, stream,
                       mask, mbits);
    hipLaunchKernelGGL((attn_kernel<true>), grid, block, 0, stream,
                       Q, K, V, mask, (const unsigned*)mbits, O);
  } else {
    hipLaunchKernelGGL((attn_kernel<false>), grid, block, 0, stream,
                       Q, K, V, mask, (const unsigned*)nullptr, O);
  }
}
